// LocalFWL_48661979464168
// MI455X (gfx1250) — matrix operand order repaired and checked by reading, not compiled, not run on hardware
//
#include <hip/hip_runtime.h>
#include <hip/hip_bf16.h>

// Problem constants (match reference)
#define N_NODES 1024
#define N_EDGES 65536
#define N_POS   1024
#define IN_DIM  128
#define H_DIM   64
#define LN_EPS  1e-5f

typedef __attribute__((ext_vector_type(16))) __bf16 bf16x16;
typedef __attribute__((ext_vector_type(8)))  __bf16 bf16x8;
typedef __attribute__((ext_vector_type(8)))  float  f32x8;
typedef int v4i __attribute__((vector_size(16)));

// gfx1250 async global->LDS staging (ASYNCcnt path), guarded so a missing
// builtin falls back to plain LDS staging without breaking the build.
#if defined(__HIP_DEVICE_COMPILE__) && \
    __has_builtin(__builtin_amdgcn_global_load_async_to_lds_b128) && \
    __has_builtin(__builtin_amdgcn_s_wait_asynccnt)
#define HAVE_ASYNC_LDS 1
#else
#define HAVE_ASYNC_LDS 0
#endif

// Builtin signature (from hipcc diagnostic): param0 is int4* in the global
// (__device__/AS1) address space; param1 is the LDS (AS3) side.
#define AS1V(p) ((__attribute__((address_space(1))) v4i*)(p))
#define AS3V(p) ((__attribute__((address_space(3))) v4i*)(p))

// ---------------------------------------------------------------------------
// Small utility kernels
// ---------------------------------------------------------------------------
__global__ void zero_i32_kernel(int* p, int n) {
    int i = blockIdx.x * blockDim.x + threadIdx.x;
    if (i < n) p[i] = 0;
}

__global__ void count_deg_kernel(const int* __restrict__ col, int* cnt, int e) {
    int i = blockIdx.x * blockDim.x + threadIdx.x;
    if (i < e) atomicAdd(&cnt[col[i]], 1);
}

__global__ void dinv_kernel(const int* __restrict__ cnt, float* dinv, int n) {
    int i = blockIdx.x * blockDim.x + threadIdx.x;
    if (i < n) dinv[i] = rsqrtf((float)cnt[i] + 1.0f);
}

__global__ void cvt_bf16_kernel(const float* __restrict__ src, __bf16* dst, int n) {
    int i = blockIdx.x * blockDim.x + threadIdx.x;
    if (i < n) dst[i] = (__bf16)src[i];
}

// ---------------------------------------------------------------------------
// WMMA GEMM: C[M x 64] = A[M x K] (bf16, row-major) @ W[K x 64] (bf16, row-major)
// Block = 128 threads = 4 wave32s; each wave computes one 16x16 tile of a
// 16-row x 64-col strip.
//  - W panel is (optionally async-)staged into LDS, then transposed to W^T with
//    a padded stride so B fragments are two contiguous 8xbf16 runs -> ds_load_b128.
//  - A fragments are two contiguous 8xbf16 runs per lane -> global_load_b128.
// EPI: 0 = store f32 raw (GCN xw)
//      1 = +bias, ReLU, store bf16 (mlp3 hidden)
//      2 = +bias, LayerNorm(gamma,beta), ReLU, store bf16 (mlp1/mlp2)
// ---------------------------------------------------------------------------
#define WT_STRIDE 136   // 128 (max K) + 8 pad; 272B row stride -> 16B aligned, bank-staggered

template <int EPI>
__launch_bounds__(128)
__global__ void gemm_wmma_kernel(const __bf16* __restrict__ A,
                                 const __bf16* __restrict__ W,
                                 int K,
                                 const float* __restrict__ bias,
                                 const float* __restrict__ gamma,
                                 const float* __restrict__ beta,
                                 float* __restrict__ outF,
                                 __bf16* __restrict__ outB) {
    __shared__ __align__(16) __bf16 sWt[H_DIM * WT_STRIDE];  // W^T: [n][k]
#if HAVE_ASYNC_LDS
    __shared__ __align__(16) __bf16 sWraw[IN_DIM * H_DIM];   // row-major stage
#endif
    __shared__ float sC[16 * H_DIM];
    __shared__ float sMu[16];
    __shared__ float sRs[16];

    const int tid  = threadIdx.x;
    const int lane = tid & 31;
    const int wave = tid >> 5;
    const int half = lane >> 4;   // wave32 half-select
    const int l16  = lane & 15;
    const int row0 = blockIdx.x * 16;
    const int n0   = wave * 16;

#if HAVE_ASYNC_LDS
    {
        // Async DMA the W panel into LDS (ASYNCcnt), then transpose in LDS.
        const char* gw = (const char*)W;
        char*       lw = (char*)sWraw;
        const int bytes = K * H_DIM * 2;
        for (int off = tid * 16; off < bytes; off += 128 * 16) {
            __builtin_amdgcn_global_load_async_to_lds_b128(
                AS1V(gw + off), AS3V(lw + off), 0, 0);
        }
        __builtin_amdgcn_s_wait_asynccnt(0);
    }
    __syncthreads();
    for (int i = tid; i < K * H_DIM; i += 128) {
        int k = i >> 6, n = i & 63;
        sWt[n * WT_STRIDE + k] = sWraw[i];
    }
#else
    for (int i = tid; i < K * H_DIM; i += 128) {
        int k = i >> 6, n = i & 63;
        sWt[n * WT_STRIDE + k] = W[i];
    }
#endif
    __syncthreads();

    f32x8 c = {};
    const __bf16* Arow = A + (size_t)(row0 + l16) * K;
    const __bf16* Bcol = sWt + (n0 + l16) * WT_STRIDE;

    for (int kk = 0; kk < K; kk += 32) {
        // A fragment (16-bit A 16x32 layout): lane holds row M=l16; its 16
        // values are two contiguous 8-element runs -> two global_load_b128.
        bf16x8 alo = *(const bf16x8*)(Arow + kk + 8 * half);
        bf16x8 ahi = *(const bf16x8*)(Arow + kk + 16 + 8 * half);
        bf16x16 a = __builtin_shufflevector(alo, ahi,
                0, 1, 2, 3, 4, 5, 6, 7, 8, 9, 10, 11, 12, 13, 14, 15);
        // B fragment (16-bit B 32x16): lane holds col N=n0+l16; its 16 K
        // values are two 8-element runs (same K order as A) -> two ds_load_b128.
        bf16x8 blo = *(const bf16x8*)(Bcol + kk + 8 * half);
        bf16x8 bhi = *(const bf16x8*)(Bcol + kk + 16 + 8 * half);
        bf16x16 b = __builtin_shufflevector(blo, bhi,
                0, 1, 2, 3, 4, 5, 6, 7, 8, 9, 10, 11, 12, 13, 14, 15);

        c = __builtin_amdgcn_wmma_f32_16x16x32_bf16(
                /*neg_a=*/false, a, /*neg_b=*/false, b,
                /*c_mod=*/(short)0, c, /*reuse_a=*/false, /*reuse_b=*/false);
    }

    const int col = n0 + l16;   // C/D layout: VGPR r -> row r + 8*half, col lane&15

    if (EPI == 0) {
#pragma unroll
        for (int r = 0; r < 8; ++r) {
            int rl = r + 8 * half;
            outF[(size_t)(row0 + rl) * H_DIM + col] = c[r];
        }
    } else if (EPI == 1) {
#pragma unroll
        for (int r = 0; r < 8; ++r) {
            int rl = r + 8 * half;
            float v = c[r] + bias[col];
            v = v > 0.f ? v : 0.f;
            outB[(size_t)(row0 + rl) * H_DIM + col] = (__bf16)v;
        }
    } else {
#pragma unroll
        for (int r = 0; r < 8; ++r) {
            int rl = r + 8 * half;
            sC[rl * H_DIM + col] = c[r] + bias[col];
        }
        __syncthreads();
        if (tid < 16) {
            float mu = 0.f;
            for (int j = 0; j < H_DIM; ++j) mu += sC[tid * H_DIM + j];
            mu *= (1.0f / H_DIM);
            float var = 0.f;
            for (int j = 0; j < H_DIM; ++j) {
                float d = sC[tid * H_DIM + j] - mu;
                var += d * d;
            }
            var *= (1.0f / H_DIM);
            sMu[tid] = mu;
            sRs[tid] = rsqrtf(var + LN_EPS);
        }
        __syncthreads();
        for (int i = tid; i < 16 * H_DIM; i += 128) {
            int rl = i >> 6, cc = i & 63;
            float v = (sC[i] - sMu[rl]) * sRs[rl] * gamma[cc] + beta[cc];
            v = v > 0.f ? v : 0.f;
            outB[(size_t)(row0 + rl) * H_DIM + cc] = (__bf16)v;
        }
    }
}

// ---------------------------------------------------------------------------
// GCN aggregation: self-loop init then edge scatter-add (D^-1/2 A D^-1/2 X W + b)
// ---------------------------------------------------------------------------
__global__ void self_init_kernel(const float* __restrict__ xw,
                                 const float* __restrict__ dinv,
                                 const float* __restrict__ b,
                                 float* h, int n) {
    int i = blockIdx.x * blockDim.x + threadIdx.x;
    if (i < n * H_DIM) {
        int node = i >> 6, j = i & 63;
        float di = dinv[node];
        h[i] = xw[i] * di * di + b[j];
    }
}

__global__ void edge_agg_kernel(const int* __restrict__ row,
                                const int* __restrict__ col,
                                const float* __restrict__ dinv,
                                const float* __restrict__ xw,
                                float* h, int e) {
    int t = blockIdx.x * blockDim.x + threadIdx.x;
    if (t < e * H_DIM) {
        int eI = t >> 6, j = t & 63;
        int r = row[eI], c = col[eI];
        float coef = dinv[r] * dinv[c];
        atomicAdd(&h[(size_t)c * H_DIM + j], coef * xw[(size_t)r * H_DIM + j]);
    }
}

// Pairwise node products at edges (bf16 out for GEMM A operand)
__global__ void edge_feat_kernel(const int* __restrict__ row,
                                 const int* __restrict__ col,
                                 const float* __restrict__ h,
                                 __bf16* xe, int e) {
    int t = blockIdx.x * blockDim.x + threadIdx.x;
    if (t < e * H_DIM) {
        int eI = t >> 6, j = t & 63;
        xe[t] = (__bf16)(h[(size_t)row[eI] * H_DIM + j] * h[(size_t)col[eI] * H_DIM + j]);
    }
}

// Pairwise node products at queries (f32)
__global__ void pos_feat_kernel(const int* __restrict__ p0,
                                const int* __restrict__ p1,
                                const float* __restrict__ h,
                                float* xx, int p) {
    int t = blockIdx.x * blockDim.x + threadIdx.x;
    if (t < p * H_DIM) {
        int pI = t >> 6, j = t & 63;
        xx[t] = h[(size_t)p0[pI] * H_DIM + j] * h[(size_t)p1[pI] * H_DIM + j];
    }
}

// Dense edge-id index: idx[r*N+c] = e+1 (0 = no edge)
__global__ void fill_idx_kernel(const int* __restrict__ row,
                                const int* __restrict__ col,
                                int* idx, int e) {
    int i = blockIdx.x * blockDim.x + threadIdx.x;
    if (i < e) idx[(size_t)row[i] * N_NODES + col[i]] = i + 1;
}

// Sparse-sparse path product:
// pv[p,h] = sum_k x2[edge(a,k),h] * x1[edge(k,b),h]
// One block (64 threads, 2 waves) per query pair; k-loop is block-uniform.
__launch_bounds__(64)
__global__ void posval_kernel(const int* __restrict__ p0,
                              const int* __restrict__ p1,
                              const int* __restrict__ idx,
                              const __bf16* __restrict__ x1,
                              const __bf16* __restrict__ x2,
                              float* pv) {
    int p = blockIdx.x;
    int j = threadIdx.x;                 // feature lane 0..63
    int a = p0[p], b = p1[p];
    const int* rowA = idx + (size_t)a * N_NODES;
    float acc = 0.f;
    for (int k = 0; k < N_NODES; ++k) {
        __builtin_prefetch(rowA + k + 64, 0, 1);   // global_prefetch_b8
        int e1 = rowA[k];                           // uniform across block
        if (e1) {
            int e2 = idx[(size_t)k * N_NODES + b];
            if (e2) {
                acc += (float)x2[(size_t)(e1 - 1) * H_DIM + j] *
                       (float)x1[(size_t)(e2 - 1) * H_DIM + j];
            }
        }
    }
    pv[(size_t)p * H_DIM + j] = acc;
}

// feat = concat([pos_val, xx]) -> bf16 [P x 2H]
__global__ void feat_kernel(const float* __restrict__ pv,
                            const float* __restrict__ xx,
                            __bf16* feat, int p) {
    int t = blockIdx.x * blockDim.x + threadIdx.x;
    if (t < p * 2 * H_DIM) {
        int pI = t >> 7, j = t & 127;
        float v = (j < H_DIM) ? pv[(size_t)pI * H_DIM + j]
                              : xx[(size_t)pI * H_DIM + (j - H_DIM)];
        feat[t] = (__bf16)v;
    }
}

// out[p] = b_b + hid[p,:] . W_b
__global__ void out_kernel(const __bf16* __restrict__ hid,
                           const float* __restrict__ Wb,
                           const float* __restrict__ bb,
                           float* out, int p) {
    int i = blockIdx.x * blockDim.x + threadIdx.x;
    if (i < p) {
        float acc = bb[0];
        for (int j = 0; j < H_DIM; ++j)
            acc += (float)hid[(size_t)i * H_DIM + j] * Wb[j];
        out[i] = acc;
    }
}

// ---------------------------------------------------------------------------
// Host-side orchestration
// ---------------------------------------------------------------------------
extern "C" void kernel_launch(void* const* d_in, const int* in_sizes, int n_in,
                              void* d_out, int out_size, void* d_ws, size_t ws_size,
                              hipStream_t stream) {
    (void)in_sizes; (void)n_in; (void)out_size; (void)ws_size;

    const float* x    = (const float*)d_in[0];
    const int*   ei   = (const int*)d_in[1];
    const int*   pos  = (const int*)d_in[2];
    const float* Wg1  = (const float*)d_in[3];
    const float* bg1  = (const float*)d_in[4];
    const float* Wg2  = (const float*)d_in[5];
    const float* bg2  = (const float*)d_in[6];
    const float* Wm1  = (const float*)d_in[7];
    const float* bm1  = (const float*)d_in[8];
    const float* gm1  = (const float*)d_in[9];
    const float* bem1 = (const float*)d_in[10];
    const float* Wm2  = (const float*)d_in[11];
    const float* bm2  = (const float*)d_in[12];
    const float* gm2  = (const float*)d_in[13];
    const float* bem2 = (const float*)d_in[14];
    const float* Wa   = (const float*)d_in[15];
    const float* ba   = (const float*)d_in[16];
    const float* Wb   = (const float*)d_in[17];
    const float* bb   = (const float*)d_in[18];

    const int* erow = ei;            // ei[0,:]
    const int* ecol = ei + N_EDGES;  // ei[1,:]
    const int* p0   = pos;
    const int* p1   = pos + N_POS;

    // Workspace carve-out (256B aligned)
    char* wp = (char*)d_ws;
    auto alloc = [&](size_t bytes) -> char* {
        char* r = wp;
        wp += (bytes + 255) & ~(size_t)255;
        return r;
    };
    float*  dinv   = (float*) alloc((size_t)N_NODES * 4);
    int*    cnt    = (int*)   alloc((size_t)N_NODES * 4);
    __bf16* x_bf   = (__bf16*)alloc((size_t)N_NODES * IN_DIM * 2);
    __bf16* Wg1b   = (__bf16*)alloc((size_t)IN_DIM * H_DIM * 2);
    __bf16* Wg2b   = (__bf16*)alloc((size_t)H_DIM * H_DIM * 2);
    __bf16* Wm1b   = (__bf16*)alloc((size_t)H_DIM * H_DIM * 2);
    __bf16* Wm2b   = (__bf16*)alloc((size_t)H_DIM * H_DIM * 2);
    __bf16* Wab    = (__bf16*)alloc((size_t)2 * H_DIM * H_DIM * 2);
    float*  xw     = (float*) alloc((size_t)N_NODES * H_DIM * 4);
    float*  h1     = (float*) alloc((size_t)N_NODES * H_DIM * 4);
    __bf16* h1b    = (__bf16*)alloc((size_t)N_NODES * H_DIM * 2);
    float*  h2     = (float*) alloc((size_t)N_NODES * H_DIM * 4);
    __bf16* xe_bf  = (__bf16*)alloc((size_t)N_EDGES * H_DIM * 2);
    __bf16* x1_bf  = (__bf16*)alloc((size_t)N_EDGES * H_DIM * 2);
    __bf16* x2_bf  = (__bf16*)alloc((size_t)N_EDGES * H_DIM * 2);
    int*    idx    = (int*)   alloc((size_t)N_NODES * N_NODES * 4);
    float*  xx     = (float*) alloc((size_t)N_POS * H_DIM * 4);
    float*  pv     = (float*) alloc((size_t)N_POS * H_DIM * 4);
    __bf16* featb  = (__bf16*)alloc((size_t)N_POS * 2 * H_DIM * 2);
    __bf16* hidb   = (__bf16*)alloc((size_t)N_POS * H_DIM * 2);

    const int B = 256;

    // Degree / normalization
    zero_i32_kernel<<<(N_NODES + B - 1) / B, B, 0, stream>>>(cnt, N_NODES);
    zero_i32_kernel<<<(N_NODES * N_NODES + B - 1) / B, B, 0, stream>>>(idx, N_NODES * N_NODES);
    count_deg_kernel<<<(N_EDGES + B - 1) / B, B, 0, stream>>>(ecol, cnt, N_EDGES);
    dinv_kernel<<<(N_NODES + B - 1) / B, B, 0, stream>>>(cnt, dinv, N_NODES);

    // bf16 conversions for WMMA operands
    cvt_bf16_kernel<<<(N_NODES * IN_DIM + B - 1) / B, B, 0, stream>>>(x, x_bf, N_NODES * IN_DIM);
    cvt_bf16_kernel<<<(IN_DIM * H_DIM + B - 1) / B, B, 0, stream>>>(Wg1, Wg1b, IN_DIM * H_DIM);
    cvt_bf16_kernel<<<(H_DIM * H_DIM + B - 1) / B, B, 0, stream>>>(Wg2, Wg2b, H_DIM * H_DIM);
    cvt_bf16_kernel<<<(H_DIM * H_DIM + B - 1) / B, B, 0, stream>>>(Wm1, Wm1b, H_DIM * H_DIM);
    cvt_bf16_kernel<<<(H_DIM * H_DIM + B - 1) / B, B, 0, stream>>>(Wm2, Wm2b, H_DIM * H_DIM);
    cvt_bf16_kernel<<<(2 * H_DIM * H_DIM + B - 1) / B, B, 0, stream>>>(Wa, Wab, 2 * H_DIM * H_DIM);

    // GCN layer 1: xw = x @ Wg1 ; h1 = scatter + self + b
    gemm_wmma_kernel<0><<<N_NODES / 16, 128, 0, stream>>>(x_bf, Wg1b, IN_DIM,
                                                          nullptr, nullptr, nullptr, xw, nullptr);
    self_init_kernel<<<(N_NODES * H_DIM + B - 1) / B, B, 0, stream>>>(xw, dinv, bg1, h1, N_NODES);
    edge_agg_kernel<<<(N_EDGES * H_DIM + B - 1) / B, B, 0, stream>>>(erow, ecol, dinv, xw, h1, N_EDGES);

    // GCN layer 2
    cvt_bf16_kernel<<<(N_NODES * H_DIM + B - 1) / B, B, 0, stream>>>(h1, h1b, N_NODES * H_DIM);
    gemm_wmma_kernel<0><<<N_NODES / 16, 128, 0, stream>>>(h1b, Wg2b, H_DIM,
                                                          nullptr, nullptr, nullptr, xw, nullptr);
    self_init_kernel<<<(N_NODES * H_DIM + B - 1) / B, B, 0, stream>>>(xw, dinv, bg2, h2, N_NODES);
    edge_agg_kernel<<<(N_EDGES * H_DIM + B - 1) / B, B, 0, stream>>>(erow, ecol, dinv, xw, h2, N_EDGES);

    // Pairwise products
    edge_feat_kernel<<<(N_EDGES * H_DIM + B - 1) / B, B, 0, stream>>>(erow, ecol, h2, xe_bf, N_EDGES);
    pos_feat_kernel<<<(N_POS * H_DIM + B - 1) / B, B, 0, stream>>>(p0, p1, h2, xx, N_POS);

    // mlp1 / mlp2: GEMM + bias + LN + ReLU fused (dominant compute, WMMA bf16)
    gemm_wmma_kernel<2><<<N_EDGES / 16, 128, 0, stream>>>(xe_bf, Wm1b, H_DIM,
                                                          bm1, gm1, bem1, nullptr, x1_bf);
    gemm_wmma_kernel<2><<<N_EDGES / 16, 128, 0, stream>>>(xe_bf, Wm2b, H_DIM,
                                                          bm2, gm2, bem2, nullptr, x2_bf);

    // Sparse path product via dense edge-id index
    fill_idx_kernel<<<(N_EDGES + B - 1) / B, B, 0, stream>>>(erow, ecol, idx, N_EDGES);
    posval_kernel<<<N_POS, 64, 0, stream>>>(p0, p1, idx, x1_bf, x2_bf, pv);

    // Final MLP: feat = [pv | xx]; hid = relu(feat @ Wa + ba); out = hid @ Wb + bb
    feat_kernel<<<(N_POS * 2 * H_DIM + B - 1) / B, B, 0, stream>>>(pv, xx, featb, N_POS);
    gemm_wmma_kernel<1><<<N_POS / 16, 128, 0, stream>>>(featb, Wab, 2 * H_DIM,
                                                        ba, nullptr, nullptr, nullptr, hidb);
    out_kernel<<<(N_POS + B - 1) / B, B, 0, stream>>>(hidb, Wb, bb, (float*)d_out, N_POS);
}
